// BaselineGAT_Edge_33303176413851
// MI455X (gfx1250) — matrix operand order repaired and checked by reading, not compiled, not run on hardware
//
#include <hip/hip_runtime.h>
#include <math.h>

// Problem constants (match reference)
#define NNODES   50000
#define NEDGES   500000
#define FDIM     128
#define EDIM     6
#define HIDC     64
#define NHEADS   4
#define HCDIM    (NHEADS * HIDC)   // 256
#define NGRAPH   256
#define NEG_SLOPE 0.2f

typedef __attribute__((ext_vector_type(16))) __bf16 v16bf;
typedef __attribute__((ext_vector_type(8)))  float  v8f;

struct Frag32B { uint4 lo, hi; };   // 32 bytes == v16bf

// ---------------------------------------------------------------------------
// f32 -> bf16 pack, 8 elements/thread, 16B stores (n must be multiple of 8)
// ---------------------------------------------------------------------------
__global__ void f32_to_bf16_kernel(const float* __restrict__ in,
                                   __bf16* __restrict__ out, long long n8) {
  long long i = (long long)blockIdx.x * blockDim.x + threadIdx.x;
  if (i >= n8) return;
  const float4* p = (const float4*)(in + i * 8);
  float4 a = p[0], b = p[1];
  __bf16 t[8] = {(__bf16)a.x, (__bf16)a.y, (__bf16)a.z, (__bf16)a.w,
                 (__bf16)b.x, (__bf16)b.y, (__bf16)b.z, (__bf16)b.w};
  *(uint4*)(out + i * 8) = *(uint4*)t;
}

// transpose + convert weights: w[K x N] f32  ->  wt[N x K] bf16
__global__ void transpose_f32_to_bf16_kernel(const float* __restrict__ w,
                                             __bf16* __restrict__ wt,
                                             int K, int N) {
  int i = blockIdx.x * blockDim.x + threadIdx.x;
  if (i >= K * N) return;
  int n = i / K;
  int k = i - n * K;
  wt[i] = (__bf16)w[(size_t)k * N + n];
}

// ---------------------------------------------------------------------------
// GEMM: C[M x N] = A[M x K] * B[K x N], bf16 operands (B pre-transposed as
// Bt[N x K]), fp32 accumulate via v_wmma_f32_16x16x32_bf16.
// One wave computes a 16(M) x 64(N) strip: 4 accumulators reuse one A frag.
// K multiple of 32, N multiple of 64, M multiple of 16.
// ---------------------------------------------------------------------------
__global__ void gemm_bf16_wmma_kernel(const __bf16* __restrict__ A,
                                      const __bf16* __restrict__ Bt,
                                      float* __restrict__ C,
                                      int M, int N, int K) {
  const int wavesPerBlock = blockDim.x >> 5;
  const int wave = blockIdx.x * wavesPerBlock + (threadIdx.x >> 5);
  const int lane = threadIdx.x & 31;
  const int tilesN4 = N >> 6;                 // 64-wide strips
  const int tilesM  = M >> 4;
  if (wave >= tilesM * tilesN4) return;       // wave-uniform: EXEC all-1s
  const int tileM  = wave / tilesN4;
  const int tileN4 = wave - tileM * tilesN4;

  const int laneLo   = lane & 15;
  const int laneHalf = lane >> 4;             // 0 or 1
  const int rowA   = (tileM << 4) + laneLo;
  const int kHalfA = laneHalf << 3;           // 0 or 8  (A: K8..15 upper lanes)
  const int kHalfB = laneHalf << 4;           // 0 or 16 (unused: B loads use kHalfA like A)
  const int colBase = tileN4 << 6;

  const __bf16* arow = A + (size_t)rowA * K;
  const __bf16* bptr0 = Bt + (size_t)(colBase + 0 * 16 + laneLo) * K;
  const __bf16* bptr1 = Bt + (size_t)(colBase + 1 * 16 + laneLo) * K;
  const __bf16* bptr2 = Bt + (size_t)(colBase + 2 * 16 + laneLo) * K;
  const __bf16* bptr3 = Bt + (size_t)(colBase + 3 * 16 + laneLo) * K;

  v8f acc0 = {}, acc1 = {}, acc2 = {}, acc3 = {};

  for (int k0 = 0; k0 < K; k0 += 32) {
    // A fragment: elems 0..7 -> K = k0+kHalfA+i ; elems 8..15 -> +16
    Frag32B af;
    af.lo = *(const uint4*)(arow + k0 + kHalfA);
    af.hi = *(const uint4*)(arow + k0 + 16 + kHalfA);
    v16bf a = __builtin_bit_cast(v16bf, af);

    // B fragments: same K order as A (elems 0..7 -> k0+kHalfA+i ; 8..15 -> +16)
    Frag32B bf;
    bf.lo = *(const uint4*)(bptr0 + k0 + kHalfA);
    bf.hi = *(const uint4*)(bptr0 + k0 + 16 + kHalfA);
    v16bf b0 = __builtin_bit_cast(v16bf, bf);
    bf.lo = *(const uint4*)(bptr1 + k0 + kHalfA);
    bf.hi = *(const uint4*)(bptr1 + k0 + 16 + kHalfA);
    v16bf b1 = __builtin_bit_cast(v16bf, bf);
    bf.lo = *(const uint4*)(bptr2 + k0 + kHalfA);
    bf.hi = *(const uint4*)(bptr2 + k0 + 16 + kHalfA);
    v16bf b2 = __builtin_bit_cast(v16bf, bf);
    bf.lo = *(const uint4*)(bptr3 + k0 + kHalfA);
    bf.hi = *(const uint4*)(bptr3 + k0 + 16 + kHalfA);
    v16bf b3 = __builtin_bit_cast(v16bf, bf);

    acc0 = __builtin_amdgcn_wmma_f32_16x16x32_bf16(false, a, false, b0,
                                                   (short)0, acc0, false, false);
    acc1 = __builtin_amdgcn_wmma_f32_16x16x32_bf16(false, a, false, b1,
                                                   (short)0, acc1, false, false);
    acc2 = __builtin_amdgcn_wmma_f32_16x16x32_bf16(false, a, false, b2,
                                                   (short)0, acc2, false, false);
    acc3 = __builtin_amdgcn_wmma_f32_16x16x32_bf16(false, a, false, b3,
                                                   (short)0, acc3, false, false);
  }

  // C/D layout: VGPR r -> M = r (lanes 0-15) / 8+r (lanes 16-31), N = lane&15
  const int rowBase = (tileM << 4) + (laneHalf << 3);
  #pragma unroll
  for (int r = 0; r < 8; ++r) {
    float* crow = C + (size_t)(rowBase + r) * N + colBase + laneLo;
    crow[0]  = acc0[r];
    crow[16] = acc1[r];
    crow[32] = acc2[r];
    crow[48] = acc3[r];
  }
}

// ---------------------------------------------------------------------------
// Helpers
// ---------------------------------------------------------------------------
__device__ __forceinline__ void atomicMaxF(float* addr, float val) {
  unsigned int* ua = (unsigned int*)addr;
  unsigned int old = *ua;
  while (__uint_as_float(old) < val) {
    unsigned int assumed = old;
    old = atomicCAS(ua, assumed, __float_as_uint(val));
    if (old == assumed) break;
  }
}

__global__ void fill_kernel(float* __restrict__ p, float v, long long n) {
  long long i = (long long)blockIdx.x * blockDim.x + threadIdx.x;
  if (i < n) p[i] = v;
}

// ---------------------------------------------------------------------------
// Edge pass 1: logits[e,h] = sum_c leaky(xs[src,h,c]+xd[dst,h,c]+ea[h,c]) * att[h,c]
// ea computed on the fly from edge_attr (ED=6) -- avoids 512MB intermediate.
// Also atomic-max into nodemax[dst,h].
// ---------------------------------------------------------------------------
__global__ void edge_logits_kernel(const float* __restrict__ xs,
                                   const float* __restrict__ xd,
                                   const int* __restrict__ src,
                                   const int* __restrict__ dst,
                                   const float* __restrict__ eattr,
                                   const float* __restrict__ w_edge,
                                   const float* __restrict__ att,
                                   float* __restrict__ logits,
                                   float* __restrict__ nodemax,
                                   int E, int H, int C) {
  int idx = blockIdx.x * blockDim.x + threadIdx.x;
  if (idx >= E * H) return;
  const int e = idx / H;
  const int h = idx - e * H;
  const int HCl = H * C;
  const int s = src[e];
  const int d = dst[e];

  float ea[EDIM];
  #pragma unroll
  for (int k = 0; k < EDIM; ++k) ea[k] = eattr[(size_t)e * EDIM + k];

  const float* xsr  = xs + (size_t)s * HCl + h * C;
  const float* xdr  = xd + (size_t)d * HCl + h * C;
  const float* attr = att + h * C;
  const float* wer  = w_edge + h * C;

  float logit = 0.f;
  #pragma unroll 4
  for (int c = 0; c < C; ++c) {
    float t = 0.f;
    #pragma unroll
    for (int k = 0; k < EDIM; ++k) t += ea[k] * wer[k * HCl + c];
    float z = xsr[c] + xdr[c] + t;
    z = (z > 0.f) ? z : NEG_SLOPE * z;
    logit += z * attr[c];
  }
  logits[idx] = logit;
  atomicMaxF(&nodemax[d * H + h], logit);
}

// ---------------------------------------------------------------------------
// Edge pass 2: w = exp(logit - max[dst]); denom[dst,h] += w
// ---------------------------------------------------------------------------
__global__ void edge_exp_kernel(float* __restrict__ logits,
                                const float* __restrict__ nodemax,
                                float* __restrict__ denom,
                                const int* __restrict__ dst,
                                int E, int H) {
  int idx = blockIdx.x * blockDim.x + threadIdx.x;
  if (idx >= E * H) return;
  const int e = idx / H;
  const int h = idx - e * H;
  const int d = dst[e];
  float m = nodemax[d * H + h];
  float w = expf(logits[idx] - m);
  logits[idx] = w;
  atomicAdd(&denom[d * H + h], w);
}

// ---------------------------------------------------------------------------
// Edge pass 3: out[dst,h,:] += xs[src,h,:] * (w / (denom[dst,h] + 1e-16))
// ---------------------------------------------------------------------------
__global__ void edge_scatter_kernel(const float* __restrict__ xs,
                                    const float* __restrict__ logits,
                                    const float* __restrict__ denom,
                                    const int* __restrict__ src,
                                    const int* __restrict__ dst,
                                    float* __restrict__ out,
                                    int E, int H, int C) {
  int idx = blockIdx.x * blockDim.x + threadIdx.x;
  if (idx >= E * H) return;
  const int e = idx / H;
  const int h = idx - e * H;
  const int HCl = H * C;
  const int s = src[e];
  const int d = dst[e];
  const float alpha = logits[idx] / (denom[d * H + h] + 1e-16f);

  const float4* xrow = (const float4*)(xs + (size_t)s * HCl + h * C);
  float* orow = out + (size_t)d * HCl + h * C;
  #pragma unroll 4
  for (int c4 = 0; c4 < C / 4; ++c4) {
    float4 v = xrow[c4];
    atomicAdd(orow + c4 * 4 + 0, v.x * alpha);
    atomicAdd(orow + c4 * 4 + 1, v.y * alpha);
    atomicAdd(orow + c4 * 4 + 2, v.z * alpha);
    atomicAdd(orow + c4 * 4 + 3, v.w * alpha);
  }
}

// out[n,j] = ELU(acc[n,j] + bias[j])
__global__ void bias_elu_kernel(const float* __restrict__ acc,
                                const float* __restrict__ bias,
                                float* __restrict__ out,
                                long long n, int HCl) {
  long long i = (long long)blockIdx.x * blockDim.x + threadIdx.x;
  if (i >= n) return;
  float v = acc[i] + bias[(int)(i % HCl)];
  out[i] = (v > 0.f) ? v : (expf(v) - 1.f);
}

// global_add_pool: g[batch[n], c] += act[n, c]
__global__ void pool_kernel(const float* __restrict__ act,
                            const int* __restrict__ batch,
                            float* __restrict__ g, int Nn, int C) {
  int i = blockIdx.x * blockDim.x + threadIdx.x;
  if (i >= Nn * C) return;
  const int n = i / C;
  const int c = i - n * C;
  atomicAdd(&g[batch[n] * C + c], act[i]);
}

// MLP head: out[b] = ELU(g[b]@fc1 + fc1_b) @ out_w + out_b
__global__ void head_kernel(const float* __restrict__ g,
                            const float* __restrict__ fc1_w,
                            const float* __restrict__ fc1_b,
                            const float* __restrict__ out_w,
                            const float* __restrict__ out_b,
                            float* __restrict__ out) {
  __shared__ float hid[HIDC];
  const int b = blockIdx.x;
  const int t = threadIdx.x;
  float acc = fc1_b[t];
  #pragma unroll 4
  for (int c = 0; c < HIDC; ++c) acc += g[b * HIDC + c] * fc1_w[c * HIDC + t];
  acc = (acc > 0.f) ? acc : (expf(acc) - 1.f);
  hid[t] = acc * out_w[t];
  __syncthreads();
  if (t == 0) {
    float s = 0.f;
    #pragma unroll 4
    for (int i = 0; i < HIDC; ++i) s += hid[i];
    out[b] = s + out_b[0];
  }
}

// ---------------------------------------------------------------------------
// Host-side orchestration
// ---------------------------------------------------------------------------
static inline int cdiv_ll(long long a, long long b) { return (int)((a + b - 1) / b); }

static void run_gat_layer(const float* actIn, int Kin,
                          const float* w_src, const float* w_dst,
                          const float* w_edge, const float* att, const float* bias,
                          int H, int C,
                          const int* srcI, const int* dstI, const float* eattr,
                          __bf16* actBf, __bf16* wtS, __bf16* wtD,
                          float* xsB, float* xdB, float* logitsB,
                          float* nmax, float* den, float* actOut,
                          hipStream_t stream) {
  const int HCo = H * C;

  // 0) one-time per-layer conversions
  long long n8 = (long long)NNODES * Kin / 8;
  f32_to_bf16_kernel<<<cdiv_ll(n8, 256), 256, 0, stream>>>(actIn, actBf, n8);
  int wcnt = Kin * HCo;
  transpose_f32_to_bf16_kernel<<<cdiv_ll(wcnt, 256), 256, 0, stream>>>(w_src, wtS, Kin, HCo);
  transpose_f32_to_bf16_kernel<<<cdiv_ll(wcnt, 256), 256, 0, stream>>>(w_dst, wtD, Kin, HCo);

  // 1) node transforms via WMMA GEMM (16x64 strip per wave)
  long long waves = (long long)(NNODES / 16) * (HCo / 64);
  int gblocks = cdiv_ll(waves, 8);  // 8 waves / 256-thread block
  gemm_bf16_wmma_kernel<<<gblocks, 256, 0, stream>>>(actBf, wtS, xsB, NNODES, HCo, Kin);
  gemm_bf16_wmma_kernel<<<gblocks, 256, 0, stream>>>(actBf, wtD, xdB, NNODES, HCo, Kin);

  // 2) segment softmax state init
  long long nh = (long long)NNODES * H;
  fill_kernel<<<cdiv_ll(nh, 256), 256, 0, stream>>>(nmax, -__builtin_inff(), nh);
  fill_kernel<<<cdiv_ll(nh, 256), 256, 0, stream>>>(den, 0.f, nh);

  // 3) edge passes
  long long eh = (long long)NEDGES * H;
  edge_logits_kernel<<<cdiv_ll(eh, 256), 256, 0, stream>>>(
      xsB, xdB, srcI, dstI, eattr, w_edge, att, logitsB, nmax, NEDGES, H, C);

  // xd no longer needed -> reuse as message accumulator
  long long nhc = (long long)NNODES * HCo;
  fill_kernel<<<cdiv_ll(nhc, 256), 256, 0, stream>>>(xdB, 0.f, nhc);

  edge_exp_kernel<<<cdiv_ll(eh, 256), 256, 0, stream>>>(
      logitsB, nmax, den, dstI, NEDGES, H);
  edge_scatter_kernel<<<cdiv_ll(eh, 256), 256, 0, stream>>>(
      xsB, logitsB, den, srcI, dstI, xdB, NEDGES, H, C);

  // 4) bias + ELU -> next activations
  bias_elu_kernel<<<cdiv_ll(nhc, 256), 256, 0, stream>>>(xdB, bias, actOut, nhc, HCo);
}

extern "C" void kernel_launch(void* const* d_in, const int* in_sizes, int n_in,
                              void* d_out, int out_size, void* d_ws, size_t ws_size,
                              hipStream_t stream) {
  (void)in_sizes; (void)n_in; (void)out_size; (void)ws_size;

  const float* x      = (const float*)d_in[0];
  const int*   eidx   = (const int*)d_in[1];
  const float* eattr  = (const float*)d_in[2];
  const int*   batch  = (const int*)d_in[3];
  const float* w_src1 = (const float*)d_in[4];
  const float* w_dst1 = (const float*)d_in[5];
  const float* w_edge1= (const float*)d_in[6];
  const float* att1   = (const float*)d_in[7];
  const float* b1     = (const float*)d_in[8];
  const float* w_src2 = (const float*)d_in[9];
  const float* w_dst2 = (const float*)d_in[10];
  const float* w_edge2= (const float*)d_in[11];
  const float* att2   = (const float*)d_in[12];
  const float* b2     = (const float*)d_in[13];
  const float* w_src3 = (const float*)d_in[14];
  const float* w_dst3 = (const float*)d_in[15];
  const float* w_edge3= (const float*)d_in[16];
  const float* att3   = (const float*)d_in[17];
  const float* b3     = (const float*)d_in[18];
  const float* fc1_w  = (const float*)d_in[19];
  const float* fc1_b  = (const float*)d_in[20];
  const float* out_w  = (const float*)d_in[21];
  const float* out_b  = (const float*)d_in[22];

  const int* srcI = eidx;            // edge_index[0, :]
  const int* dstI = eidx + NEDGES;   // edge_index[1, :]

  // workspace carve-out (~190 MB)
  char* ws = (char*)d_ws;
  size_t off = 0;
  auto carve = [&](size_t bytes) -> char* {
    char* p = ws + off;
    off += (bytes + 255) & ~(size_t)255;
    return p;
  };
  float*  act     = (float*)carve(sizeof(float) * (size_t)NNODES * HCDIM);
  float*  xsB     = (float*)carve(sizeof(float) * (size_t)NNODES * HCDIM);
  float*  xdB     = (float*)carve(sizeof(float) * (size_t)NNODES * HCDIM);
  float*  logitsB = (float*)carve(sizeof(float) * (size_t)NEDGES * NHEADS);
  float*  nmax    = (float*)carve(sizeof(float) * (size_t)NNODES * NHEADS);
  float*  den     = (float*)carve(sizeof(float) * (size_t)NNODES * NHEADS);
  float*  g       = (float*)carve(sizeof(float) * (size_t)NGRAPH * HIDC);
  __bf16* actBf   = (__bf16*)carve(sizeof(__bf16) * (size_t)NNODES * HCDIM);
  __bf16* wtS     = (__bf16*)carve(sizeof(__bf16) * (size_t)HCDIM * HCDIM);
  __bf16* wtD     = (__bf16*)carve(sizeof(__bf16) * (size_t)HCDIM * HCDIM);

  // layer 1: in = x (N x 128), H=4, C=64, concat
  run_gat_layer(x, FDIM, w_src1, w_dst1, w_edge1, att1, b1, NHEADS, HIDC,
                srcI, dstI, eattr, actBf, wtS, wtD,
                xsB, xdB, logitsB, nmax, den, act, stream);
  // layer 2: in = act (N x 256), H=4, C=64, concat
  run_gat_layer(act, HCDIM, w_src2, w_dst2, w_edge2, att2, b2, NHEADS, HIDC,
                srcI, dstI, eattr, actBf, wtS, wtD,
                xsB, xdB, logitsB, nmax, den, act, stream);
  // layer 3: in = act (N x 256), H=1, C=64 (mean over 1 head == identity)
  run_gat_layer(act, HCDIM, w_src3, w_dst3, w_edge3, att3, b3, 1, HIDC,
                srcI, dstI, eattr, actBf, wtS, wtD,
                xsB, xdB, logitsB, nmax, den, act, stream);

  // global add pool
  long long gn = (long long)NGRAPH * HIDC;
  fill_kernel<<<cdiv_ll(gn, 256), 256, 0, stream>>>(g, 0.f, gn);
  long long nc = (long long)NNODES * HIDC;
  pool_kernel<<<cdiv_ll(nc, 256), 256, 0, stream>>>(act, batch, g, NNODES, HIDC);

  // MLP head -> d_out [G, 1]
  head_kernel<<<NGRAPH, HIDC, 0, stream>>>(g, fc1_w, fc1_b, out_w, out_b, (float*)d_out);
}
